// NaiveAttentionModule_63118839382048
// MI455X (gfx1250) — compile-verified, matrix operand order checked by reading, not run on hardware
//
#include <hip/hip_runtime.h>
#include <hip/hip_bf16.h>

// ---------------------------------------------------------------------------
// Problem constants: x [1,4096,1024] fp32, W [1024,3072] fp32 -> out fp32.
// HEADS=16, HEAD_DIM=64.
// ---------------------------------------------------------------------------
#define T_SEQ    4096
#define HIDDEN   1024
#define HEADS    16
#define HEAD_DIM 64
#define N_QKV    (3 * HIDDEN)

typedef __bf16 bf16_t;
typedef __attribute__((ext_vector_type(16))) __bf16        v16bf;
typedef __attribute__((ext_vector_type(8)))  float         v8f;
typedef __attribute__((ext_vector_type(4)))  unsigned int  v4u;

// Round-to-nearest-even fp32 -> bf16
static __device__ __forceinline__ unsigned short f32_to_bf16_bits(float f) {
  union { float f; unsigned u; } v; v.f = f;
  unsigned r = v.u + 0x7FFFu + ((v.u >> 16) & 1u);
  return (unsigned short)(r >> 16);
}
static __device__ __forceinline__ bf16_t to_bf16(float f) {
  union { unsigned short s; bf16_t b; } o; o.s = f32_to_bf16_bits(f);
  return o.b;
}

static __device__ __forceinline__ v8f wmma_bf16(v16bf a, v16bf b, v8f c) {
  return __builtin_amdgcn_wmma_f32_16x16x32_bf16(
      false, a, false, b, (short)0, c, false, false);
}

// 16x32 bf16 fragment in the CDNA5 A/B VGPR layout from row-major [rows][ldk].
//   lanes 0-15 : row = r0+lane,    K chunks {0..7} and {16..23}
//   lanes 16-31: row = r0+lane-16, K chunks {8..15} and {24..31}
static __device__ __forceinline__ v16bf load_frag16(const bf16_t* base,
                                                    int r0, int k0, int ldk) {
  const int lane = threadIdx.x & 31;
  const int row  = r0 + (lane & 15);
  const int hi   = lane >> 4;
  const bf16_t* p = base + (size_t)row * ldk + k0 + hi * 8;
  union { v16bf v; v4u q[2]; } f;
  f.q[0] = *(const v4u*)(p);
  f.q[1] = *(const v4u*)(p + 16);
  return f.v;
}

// Async global->LDS 16B per lane (GLOBAL_LOAD_ASYNC_TO_LDS_B128, GVS mode).
static __device__ __forceinline__ void async_ld_b128(unsigned lds_addr,
                                                     unsigned gl_off,
                                                     unsigned long long base) {
  asm volatile("global_load_async_to_lds_b128 %0, %1, %2"
               :: "v"(lds_addr), "v"(gl_off), "s"(base) : "memory");
}
static __device__ __forceinline__ void wait_async(int n) {
  if (n == 0) asm volatile("s_wait_asynccnt 0x0" ::: "memory");
  else if (n == 1) asm volatile("s_wait_asynccnt 0x1" ::: "memory");
  else asm volatile("s_wait_asynccnt 0x2" ::: "memory");
}
static __device__ __forceinline__ void wait_ds0() {
  asm volatile("s_wait_dscnt 0x0" ::: "memory");
}
static __device__ __forceinline__ unsigned lds_addr32(const void* p) {
  // Generic pointers to LDS carry the byte offset in addr[31:0] (ISA 10.2).
  return (unsigned)(uintptr_t)p;
}

// ---------------------------------------------------------------------------
// Kernel 1: x fp32 -> bf16
// ---------------------------------------------------------------------------
__global__ void cvt_x_kernel(const float* __restrict__ x, bf16_t* __restrict__ y, int n) {
  int i = blockIdx.x * blockDim.x + threadIdx.x;
  if (i < n) y[i] = to_bf16(x[i]);
}

// ---------------------------------------------------------------------------
// Kernel 2: W [K=1024][N=3072] fp32 -> Wt [N=3072][K=1024] bf16 (transpose)
// ---------------------------------------------------------------------------
__global__ void cvt_wt_kernel(const float* __restrict__ w, bf16_t* __restrict__ wt) {
  int i = blockIdx.x * blockDim.x + threadIdx.x;
  int k = i & (HIDDEN - 1);
  int n = i >> 10;
  wt[i] = to_bf16(w[(size_t)k * N_QKV + n]);
}

// ---------------------------------------------------------------------------
// Kernel 3: QKV GEMM.  Block = 8 waves = 256 rows x 64 cols; each wave owns a
// 32x64 tile (2 A-frags x 4 B-frags = 8 WMMAs/k-step, B reused across rows).
// Wt tile (64n x 32k = 4KB) async-staged to LDS per k-step, double buffered;
// A fragments register-pipelined one step ahead.
// ---------------------------------------------------------------------------
__global__ __launch_bounds__(256) void qkv_gemm_kernel(
    const bf16_t* __restrict__ X, const bf16_t* __restrict__ Wt,
    bf16_t* __restrict__ Qh, bf16_t* __restrict__ Kh, bf16_t* __restrict__ Vt) {
  __shared__ __align__(16) bf16_t btile[2][64 * 32];   // [n:64][k:32], 4KB each

  const int tid  = threadIdx.x;
  const int wib  = tid >> 5;
  const int lane = tid & 31;
  const int hi   = lane >> 4;
  const int col  = lane & 15;
  const int nst  = blockIdx.x % (N_QKV / 64);          // 48 supertiles
  const int t0   = (blockIdx.x / (N_QKV / 64)) * 256 + wib * 32;
  const int n0   = nst * 64;

  // Thread tid copies 16B of the [64][32] tile: n-row = tid/4, k-col = (tid%4)*8
  const unsigned srcElem = (unsigned)((n0 + (tid >> 2)) * HIDDEN + (tid & 3) * 8);
  const unsigned long long wbase = (unsigned long long)(uintptr_t)Wt;
  const unsigned lds0 = lds_addr32(&btile[0][0]) + (unsigned)(tid * 16);
  const unsigned lds1 = lds_addr32(&btile[1][0]) + (unsigned)(tid * 16);

  async_ld_b128(lds0, srcElem * 2u, wbase);            // stage k0 = 0

  v16bf a0 = load_frag16(X, t0,      0, HIDDEN);
  v16bf a1 = load_frag16(X, t0 + 16, 0, HIDDEN);
  v8f acc00 = {}, acc01 = {}, acc02 = {}, acc03 = {};
  v8f acc10 = {}, acc11 = {}, acc12 = {}, acc13 = {};

  for (int k0 = 0; k0 < HIDDEN; k0 += 32) {
    const int cur = (k0 >> 5) & 1;
    const bf16_t* bb = btile[cur];
    if (k0 + 32 < HIDDEN) {
      async_ld_b128(cur ? lds0 : lds1, (srcElem + (unsigned)(k0 + 32)) * 2u, wbase);
      wait_async(1);                                   // current buffer landed
    } else {
      wait_async(0);
    }
    __syncthreads();

    const v16bf ca0 = a0, ca1 = a1;
    if (k0 + 32 < HIDDEN) {                            // pipeline A one step
      a0 = load_frag16(X, t0,      k0 + 32, HIDDEN);
      a1 = load_frag16(X, t0 + 16, k0 + 32, HIDDEN);
    }

    const v16bf b0 = load_frag16(bb, 0,  0, 32);
    const v16bf b1 = load_frag16(bb, 16, 0, 32);
    const v16bf b2 = load_frag16(bb, 32, 0, 32);
    const v16bf b3 = load_frag16(bb, 48, 0, 32);
    acc00 = wmma_bf16(ca0, b0, acc00);
    acc10 = wmma_bf16(ca1, b0, acc10);
    acc01 = wmma_bf16(ca0, b1, acc01);
    acc11 = wmma_bf16(ca1, b1, acc11);
    acc02 = wmma_bf16(ca0, b2, acc02);
    acc12 = wmma_bf16(ca1, b2, acc12);
    acc03 = wmma_bf16(ca0, b3, acc03);
    acc13 = wmma_bf16(ca1, b3, acc13);

    wait_ds0();
    __syncthreads();                                   // safe to overwrite buf
  }

  const int s = n0 / HIDDEN;                           // 0=Q,1=K,2=V (uniform)
  const int h = (n0 % HIDDEN) / HEAD_DIM;
  v8f accs[2][4] = {{acc00, acc01, acc02, acc03}, {acc10, acc11, acc12, acc13}};
#pragma unroll
  for (int r = 0; r < 2; ++r) {
#pragma unroll
    for (int f = 0; f < 4; ++f) {
      const int d = f * 16 + col;
#pragma unroll
      for (int j = 0; j < 8; ++j) {
        const int t = t0 + r * 16 + j + hi * 8;
        bf16_t val = to_bf16(accs[r][f][j]);
        if (s == 0)      Qh[((size_t)(h * T_SEQ + t)) * HEAD_DIM + d] = val;
        else if (s == 1) Kh[((size_t)(h * T_SEQ + t)) * HEAD_DIM + d] = val;
        else             Vt[((size_t)(h * HEAD_DIM + d)) * T_SEQ + t] = val;
      }
    }
  }
}

// ---------------------------------------------------------------------------
// Kernel 4: flash attention.  Block = 8 waves = one head, 128 queries.
// 32-key K/V tiles async-staged to LDS once per block, double buffered.
// ---------------------------------------------------------------------------
__global__ __launch_bounds__(256) void flash_attn_kernel(
    const bf16_t* __restrict__ Qh, const bf16_t* __restrict__ Kh,
    const bf16_t* __restrict__ Vt, float* __restrict__ out) {
  __shared__ __align__(16) bf16_t ktile[2][32 * 64];   // [key][dim] 4KB
  __shared__ __align__(16) bf16_t vtile[2][64 * 32];   // [dim][key] 4KB
  __shared__ __align__(16) bf16_t ptile[8][16 * 32];   // per-wave P

  const int tid  = threadIdx.x;
  const int wib  = tid >> 5;
  const int lane = tid & 31;
  const int hi   = lane >> 4;
  const int col  = lane & 15;
  const int h    = blockIdx.x / (T_SEQ / 128);         // 32 blocks per head
  const int q0   = (blockIdx.x % (T_SEQ / 128)) * 128 + wib * 16;

  const bf16_t* Qp = Qh + (size_t)h * T_SEQ * HEAD_DIM;
  const bf16_t* Kp = Kh + (size_t)h * T_SEQ * HEAD_DIM;
  const bf16_t* Vp = Vt + (size_t)h * HEAD_DIM * T_SEQ;
  bf16_t* myt = ptile[wib];

  const unsigned long long kbase = (unsigned long long)(uintptr_t)Kp;
  const unsigned long long vbase = (unsigned long long)(uintptr_t)Vp;
  // K tile: contiguous 4KB block (rows kt..kt+31 of [T][64]); tid copies 16B.
  const unsigned kSrc = (unsigned)(tid * 8);           // + kt*64, elements
  // V tile: dim = tid/4, keys (tid%4)*8..+7 from [64][T]; LDS layout [dim][32].
  const unsigned vSrc = (unsigned)((tid >> 2) * T_SEQ + (tid & 3) * 8);  // + kt
  unsigned klds[2], vlds[2];
  klds[0] = lds_addr32(&ktile[0][0]) + (unsigned)(tid * 16);
  klds[1] = lds_addr32(&ktile[1][0]) + (unsigned)(tid * 16);
  vlds[0] = lds_addr32(&vtile[0][0]) + (unsigned)(tid * 16);
  vlds[1] = lds_addr32(&vtile[1][0]) + (unsigned)(tid * 16);

  // Prologue: stage kt = 0 into buffer 0 (2 async ops / wave).
  async_ld_b128(klds[0], kSrc * 2u, kbase);
  async_ld_b128(vlds[0], vSrc * 2u, vbase);

  const v16bf aQ0 = load_frag16(Qp, q0, 0,  HEAD_DIM);
  const v16bf aQ1 = load_frag16(Qp, q0, 32, HEAD_DIM);

  v8f o0 = {}, o1 = {}, o2 = {}, o3 = {};
  float mrow[8], lrow[8];
#pragma unroll
  for (int j = 0; j < 8; ++j) { mrow[j] = -__builtin_inff(); lrow[j] = 0.0f; }

  const float scale = 0.125f;                          // 1/sqrt(64)

  for (int kt = 0; kt < T_SEQ; kt += 32) {
    const int cur = (kt >> 5) & 1;
    const bf16_t* kb = ktile[cur];
    const bf16_t* vb = vtile[cur];
    if (kt + 32 < T_SEQ) {                             // prefetch next tiles
      const int nxt = cur ^ 1;
      async_ld_b128(klds[nxt], (kSrc + (unsigned)(kt + 32) * 64u) * 2u, kbase);
      async_ld_b128(vlds[nxt], (vSrc + (unsigned)(kt + 32)) * 2u, vbase);
      wait_async(2);                                   // current tiles landed
    } else {
      wait_async(0);
    }
    __syncthreads();

    // S = Q K^T for keys [kt, kt+32) -- B fragments from LDS K tile [32][64]
    const v16bf bk00 = load_frag16(kb, 0,  0,  HEAD_DIM);
    const v16bf bk01 = load_frag16(kb, 0,  32, HEAD_DIM);
    const v16bf bk10 = load_frag16(kb, 16, 0,  HEAD_DIM);
    const v16bf bk11 = load_frag16(kb, 16, 32, HEAD_DIM);
    v8f s0 = {}, s1 = {};
    s0 = wmma_bf16(aQ0, bk00, s0);
    s0 = wmma_bf16(aQ1, bk01, s0);
    s1 = wmma_bf16(aQ0, bk10, s1);
    s1 = wmma_bf16(aQ1, bk11, s1);

    // Online softmax: C layout -> each row in one 16-lane half.
#pragma unroll
    for (int j = 0; j < 8; ++j) {
      float x0 = s0[j] * scale;
      float x1 = s1[j] * scale;
      float mx = fmaxf(x0, x1);
#pragma unroll
      for (int m = 1; m < 16; m <<= 1) mx = fmaxf(mx, __shfl_xor(mx, m, 16));
      float newm = fmaxf(mrow[j], mx);
      float corr = __expf(mrow[j] - newm);
      float p0 = __expf(x0 - newm);
      float p1 = __expf(x1 - newm);
      float rs = p0 + p1;
#pragma unroll
      for (int m = 1; m < 16; m <<= 1) rs += __shfl_xor(rs, m, 16);
      lrow[j] = lrow[j] * corr + rs;
      mrow[j] = newm;
      o0[j] *= corr; o1[j] *= corr; o2[j] *= corr; o3[j] *= corr;
      const int mr = j + hi * 8;
      myt[mr * 32 + col]      = to_bf16(p0);
      myt[mr * 32 + 16 + col] = to_bf16(p1);
    }
    wait_ds0();                                        // cross-lane LDS RAW

    // P as 16x32 A fragment (LDS transpose of C layout)
    union { v16bf v; v4u q[2]; } fp;
    const bf16_t* pr = myt + (size_t)col * 32 + hi * 8;
    fp.q[0] = *(const v4u*)(pr);
    fp.q[1] = *(const v4u*)(pr + 16);
    const v16bf aP = fp.v;

    // O += P V -- B fragments from LDS V tile [64][32]
    const v16bf bv0 = load_frag16(vb, 0,  0, 32);
    const v16bf bv1 = load_frag16(vb, 16, 0, 32);
    const v16bf bv2 = load_frag16(vb, 32, 0, 32);
    const v16bf bv3 = load_frag16(vb, 48, 0, 32);
    o0 = wmma_bf16(aP, bv0, o0);
    o1 = wmma_bf16(aP, bv1, o1);
    o2 = wmma_bf16(aP, bv2, o2);
    o3 = wmma_bf16(aP, bv3, o3);

    wait_ds0();
    __syncthreads();                                   // buffer may be reused
  }

#pragma unroll
  for (int j = 0; j < 8; ++j) {
    const int t = q0 + j + hi * 8;
    const float inv = 1.0f / lrow[j];
    float* orow = out + (size_t)t * HIDDEN + h * HEAD_DIM;
    orow[ 0 + col] = o0[j] * inv;
    orow[16 + col] = o1[j] * inv;
    orow[32 + col] = o2[j] * inv;
    orow[48 + col] = o3[j] * inv;
  }
}

// ---------------------------------------------------------------------------
// Workspace layout (bytes):
//   [0,        8 MiB)  X16  bf16 [4096][1024]
//   [8 MiB,   14 MiB)  Wt16 bf16 [3072][1024]   (W transposed)
//   [14 MiB,  22 MiB)  Qh   bf16 [16][4096][64]
//   [22 MiB,  30 MiB)  Kh   bf16 [16][4096][64]
//   [30 MiB,  38 MiB)  Vt   bf16 [16][64][4096]
// ---------------------------------------------------------------------------
extern "C" void kernel_launch(void* const* d_in, const int* in_sizes, int n_in,
                              void* d_out, int out_size, void* d_ws, size_t ws_size,
                              hipStream_t stream) {
  (void)in_sizes; (void)n_in; (void)out_size; (void)ws_size;
  const float* x = (const float*)d_in[0];
  const float* w = (const float*)d_in[1];
  float* out = (float*)d_out;
  char* ws = (char*)d_ws;

  bf16_t* X16  = (bf16_t*)(ws);
  bf16_t* Wt16 = (bf16_t*)(ws + (size_t)8  * 1024 * 1024);
  bf16_t* Qh   = (bf16_t*)(ws + (size_t)14 * 1024 * 1024);
  bf16_t* Kh   = (bf16_t*)(ws + (size_t)22 * 1024 * 1024);
  bf16_t* Vt   = (bf16_t*)(ws + (size_t)30 * 1024 * 1024);

  cvt_x_kernel<<<(T_SEQ * HIDDEN) / 256, 256, 0, stream>>>(x, X16, T_SEQ * HIDDEN);
  cvt_wt_kernel<<<(HIDDEN * N_QKV) / 256, 256, 0, stream>>>(w, Wt16);

  // 16 t-blocks (256 rows each) * 48 n-supertiles
  qkv_gemm_kernel<<<16 * 48, 256, 0, stream>>>(X16, Wt16, Qh, Kh, Vt);

  // 16 heads * 32 blocks, each block = 128 queries
  flash_attn_kernel<<<HEADS * (T_SEQ / 128), 256, 0, stream>>>(Qh, Kh, Vt, out);
}
